// Decoder_51582557225714
// MI455X (gfx1250) — matrix operand order repaired and checked by reading, not compiled, not run on hardware
//
#include <hip/hip_runtime.h>
#include <math.h>

// ---------------------------------------------------------------------------
// Decoder block for MI455X (gfx1250).
//   * Weights transposed+converted once to bf16 (rotating scratch slot).
//   * Activations flow in bf16; fp32 residual stream; fp32 WMMA accumulate.
//   * GEMM: 128x128 block tile, 8 wave32 waves, 32x64 wave tile -> 8 WMMA per
//     K-step; double-buffered LDS staged via async global->LDS (ASYNCcnt)
//     when the toolchain exposes the gfx1250 builtins, ushort8 copies else.
//   * Flash attention on WMMA with direct bf16 fragment loads.
// ---------------------------------------------------------------------------

#define S_LEN 1024
#define D_MODEL 4096
#define N_HEADS 32
#define HEAD_DIM 128
#define FF_DIM 11008
#define VOCAB 8000
#define VOCAB_PAD 8064          // round up to 128
#define DM_MEM 1024
#define PH_DIM 8192

typedef __attribute__((ext_vector_type(16))) __bf16 v16bf;
typedef __attribute__((ext_vector_type(8)))  float  v8f;
typedef __attribute__((ext_vector_type(8)))  unsigned short ushort8;
typedef __attribute__((vector_size(16)))     int    v4i;   // <4 x i32> for async builtin

union Frag {
    v16bf v;
    ushort8 h[2];
    unsigned short s[16];
};

__device__ __forceinline__ unsigned short f32_to_bf16(float f) {
    unsigned int u = __float_as_uint(f);
    u += 0x7FFFu + ((u >> 16) & 1u);   // round-to-nearest-even
    return (unsigned short)(u >> 16);
}
__device__ __forceinline__ float bf16_to_f32(unsigned short u) {
    return __uint_as_float((unsigned int)u << 16);
}

// ---- gfx1250 async global->LDS staging (guarded; fallback = plain copy) ----
#if __has_builtin(__builtin_amdgcn_global_load_async_to_lds_b128)
#define ASYNC_LDS 1
#endif

__device__ __forceinline__ void copy16(const unsigned short* g, unsigned short* l) {
#ifdef ASYNC_LDS
    __builtin_amdgcn_global_load_async_to_lds_b128((v4i*)g, (v4i*)l, 0, 0);
#else
    *(ushort8*)l = *(const ushort8*)g;
#endif
}
__device__ __forceinline__ void wait_async_lds() {
#ifdef ASYNC_LDS
#if __has_builtin(__builtin_amdgcn_s_wait_asynccnt)
    __builtin_amdgcn_s_wait_asynccnt(0);
#else
    asm volatile("s_wait_asynccnt 0x0" ::: "memory");
#endif
#endif
}

// act: 0 = none, 1 = exact GELU, 2 = SiLU
template <int ACT>
__device__ __forceinline__ float apply_act(float x) {
    if (ACT == 1) return 0.5f * x * (1.0f + erff(x * 0.70710678118654752f));
    if (ACT == 2) return x / (1.0f + __expf(-x));
    return x;
}

// ---------------------------------------------------------------------------
// GEMM: C[M,N] = epilogue(A[M,K] @ Bt[N,K]^T).  A,Bt bf16; C fp32 or bf16.
// M%128==0, K%32==0, Npad%128==0 (Bt rows zero-padded to Npad).
// ---------------------------------------------------------------------------
template <int ACT, bool OBF16>
__global__ __launch_bounds__(256)
void gemm_bf16_wmma(const unsigned short* __restrict__ A,
                    const unsigned short* __restrict__ Bt,
                    const float* __restrict__ bias, const float* __restrict__ resid,
                    void* __restrict__ Cv, int M, int N, int K) {
    __shared__ __align__(16) unsigned short As[2][128][40];  // [m][k]
    __shared__ __align__(16) unsigned short Bs[2][128][40];  // [n][k]

    const int tid  = threadIdx.x;
    const int wave = tid >> 5;
    const int lane = tid & 31;
    const int col0 = blockIdx.x * 128;
    const int row0 = blockIdx.y * 128;

    const int wm = wave & 3;   // 4 row groups of 32
    const int wn = wave >> 2;  // 2 col groups of 64
    const int m16 = lane & 15;
    const int kh  = lane >> 4;

    v8f acc[2][4] = {};

    auto stage = [&](int k0, int buf) {
        #pragma unroll
        for (int i = 0; i < 2; ++i) {               // A tile: 512 x 16B chunks
            int idx = i * 256 + tid;
            int r = idx >> 2, c8 = idx & 3;
            copy16(&A[(size_t)(row0 + r) * K + k0 + c8 * 8], &As[buf][r][c8 * 8]);
        }
        #pragma unroll
        for (int i = 0; i < 2; ++i) {               // B tile: 512 x 16B chunks
            int idx = i * 256 + tid;
            int r = idx >> 2, c8 = idx & 3;
            copy16(&Bt[(size_t)(col0 + r) * K + k0 + c8 * 8], &Bs[buf][r][c8 * 8]);
        }
    };

    stage(0, 0);
    int buf = 0;
    for (int k0 = 0; k0 < K; k0 += 32) {
        wait_async_lds();
        __syncthreads();
        if (k0 + 32 < K) stage(k0 + 32, buf ^ 1);   // overlap with WMMAs below

        Frag fa[2], fb[4];
        #pragma unroll
        for (int i = 0; i < 2; ++i) {
            int r = wm * 32 + i * 16 + m16;
            fa[i].h[0] = *(const ushort8*)&As[buf][r][kh * 8];
            fa[i].h[1] = *(const ushort8*)&As[buf][r][16 + kh * 8];
        }
        #pragma unroll
        for (int j = 0; j < 4; ++j) {
            int n = wn * 64 + j * 16 + m16;
            fb[j].h[0] = *(const ushort8*)&Bs[buf][n][kh * 8];
            fb[j].h[1] = *(const ushort8*)&Bs[buf][n][16 + kh * 8];
        }
        #pragma unroll
        for (int i = 0; i < 2; ++i)
            #pragma unroll
            for (int j = 0; j < 4; ++j)
                acc[i][j] = __builtin_amdgcn_wmma_f32_16x16x32_bf16(
                    false, fa[i].v, false, fb[j].v, (short)0, acc[i][j],
                    false, false);
        buf ^= 1;
    }

    const int nl = lane & 15;
    const int mh = lane >> 4;
    #pragma unroll
    for (int i = 0; i < 2; ++i) {
        #pragma unroll
        for (int j = 0; j < 4; ++j) {
            int ng = col0 + wn * 64 + j * 16 + nl;
            if (ng < N) {
                #pragma unroll
                for (int r = 0; r < 8; ++r) {
                    int mg = row0 + wm * 32 + i * 16 + mh * 8 + r;
                    float v = acc[i][j][r];
                    if (bias) v += bias[ng];
                    v = apply_act<ACT>(v);
                    size_t off = (size_t)mg * N + ng;
                    if (OBF16) {
                        ((unsigned short*)Cv)[off] = f32_to_bf16(v);
                    } else {
                        if (resid) v += resid[off];
                        ((float*)Cv)[off] = v;
                    }
                }
            }
        }
    }
}

// ---------------------------------------------------------------------------
// WMMA flash attention (causal), all operands bf16.  grid (S/32, H),
// 64 threads = 2 waves, each wave owns a 16-query tile, 32-key blocks.
// ---------------------------------------------------------------------------
__global__ __launch_bounds__(64)
void attn_wmma_kernel(const unsigned short* __restrict__ Q,
                      const unsigned short* __restrict__ K,
                      const unsigned short* __restrict__ V,
                      unsigned short* __restrict__ O) {
    __shared__ __align__(16) unsigned short Vt[2][128][40];  // [dh][key] per wave
    __shared__ __align__(16) unsigned short Ps[2][16][40];   // [q][key]  per wave

    const int wave = threadIdx.x >> 5;
    const int lane = threadIdx.x & 31;
    const int head = blockIdx.y;
    const int q0   = blockIdx.x * 32 + wave * 16;
    const int n16  = lane & 15;
    const int kh   = lane >> 4;
    const float scale = 0.08838834764831845f;  // 1/sqrt(128)

    // ---- Q tile resident as 4 A-fragments: direct 16B loads ----
    Frag aq[4];
    {
        const unsigned short* qr = Q + (size_t)(q0 + n16) * D_MODEL + head * HEAD_DIM;
        #pragma unroll
        for (int c = 0; c < 4; ++c) {
            aq[c].h[0] = *(const ushort8*)(qr + c * 32 + kh * 8);
            aq[c].h[1] = *(const ushort8*)(qr + c * 32 + 16 + kh * 8);
        }
    }

    float mrow[8], lrow[8];
    v8f ofrag[8] = {};
    #pragma unroll
    for (int r = 0; r < 8; ++r) { mrow[r] = -3.4e38f; lrow[r] = 0.f; }

    const int nblocks = q0 / 32 + 1;
    for (int kb = 0; kb < nblocks; ++kb) {
        const int kbase = kb * 32;

        // ---- stage V tile transposed into LDS: Vt[dh][key] ----
        #pragma unroll 4
        for (int key = 0; key < 32; ++key) {
            const unsigned short* vr =
                V + (size_t)(kbase + key) * D_MODEL + head * HEAD_DIM + lane * 4;
            Vt[wave][lane * 4 + 0][key] = vr[0];
            Vt[wave][lane * 4 + 1][key] = vr[1];
            Vt[wave][lane * 4 + 2][key] = vr[2];
            Vt[wave][lane * 4 + 3][key] = vr[3];
        }

        // ---- K fragments: direct 16B loads from global (L2-resident) ----
        Frag bk[2][4];
        #pragma unroll
        for (int t = 0; t < 2; ++t) {
            const unsigned short* kr =
                K + (size_t)(kbase + t * 16 + n16) * D_MODEL + head * HEAD_DIM;
            #pragma unroll
            for (int c = 0; c < 4; ++c) {
                bk[t][c].h[0] = *(const ushort8*)(kr + c * 32 + kh * 8);
                bk[t][c].h[1] = *(const ushort8*)(kr + c * 32 + 16 + kh * 8);
            }
        }

        // ---- S = Q @ K^T ----
        v8f s0 = {}, s1 = {};
        #pragma unroll
        for (int c = 0; c < 4; ++c) {
            s0 = __builtin_amdgcn_wmma_f32_16x16x32_bf16(false, aq[c].v, false, bk[0][c].v,
                                                         (short)0, s0, false, false);
            s1 = __builtin_amdgcn_wmma_f32_16x16x32_bf16(false, aq[c].v, false, bk[1][c].v,
                                                         (short)0, s1, false, false);
        }

        // ---- online softmax in C-layout: lane = key, (r, lane>>4) = query ----
        float corr[8];
        #pragma unroll
        for (int r = 0; r < 8; ++r) {
            int qg = q0 + r + 8 * kh;
            float a = s0[r] * scale; if (kbase + n16 > qg)      a = -3.4e38f;
            float b = s1[r] * scale; if (kbase + 16 + n16 > qg) b = -3.4e38f;
            float mx = fmaxf(a, b);
            #pragma unroll
            for (int off = 8; off > 0; off >>= 1) mx = fmaxf(mx, __shfl_xor(mx, off, 32));
            float mn = fmaxf(mrow[r], mx);
            corr[r] = __expf(mrow[r] - mn);
            float p0 = __expf(a - mn);
            float p1 = __expf(b - mn);
            float rs = p0 + p1;
            #pragma unroll
            for (int off = 8; off > 0; off >>= 1) rs += __shfl_xor(rs, off, 32);
            lrow[r] = lrow[r] * corr[r] + rs;
            mrow[r] = mn;
            int m = r + 8 * kh;                  // transpose P through LDS
            Ps[wave][m][n16]      = f32_to_bf16(p0);
            Ps[wave][m][16 + n16] = f32_to_bf16(p1);
        }

        Frag pa;
        pa.h[0] = *(const ushort8*)&Ps[wave][n16][kh * 8];
        pa.h[1] = *(const ushort8*)&Ps[wave][n16][16 + kh * 8];

        // ---- O += P @ V ----
        #pragma unroll
        for (int f = 0; f < 8; ++f) {
            Frag vb;
            vb.h[0] = *(const ushort8*)&Vt[wave][f * 16 + n16][kh * 8];
            vb.h[1] = *(const ushort8*)&Vt[wave][f * 16 + n16][16 + kh * 8];
            #pragma unroll
            for (int r = 0; r < 8; ++r) ofrag[f][r] *= corr[r];
            ofrag[f] = __builtin_amdgcn_wmma_f32_16x16x32_bf16(false, pa.v, false, vb.v,
                                                               (short)0, ofrag[f],
                                                               false, false);
        }
    }

    #pragma unroll
    for (int f = 0; f < 8; ++f) {
        #pragma unroll
        for (int r = 0; r < 8; ++r) {
            int m = r + 8 * kh;
            O[(size_t)(q0 + m) * D_MODEL + head * HEAD_DIM + f * 16 + n16] =
                f32_to_bf16(ofrag[f][r] / lrow[r]);
        }
    }
}

// ---------------------------------------------------------------------------
// Weight transpose + convert: src fp32 [K,N] -> dst bf16 [Npad,K], zero-pad.
// grid (K/32, Npad/32), 256 threads (32x8).
// ---------------------------------------------------------------------------
__global__ __launch_bounds__(256)
void cvt_wT_kernel(const float* __restrict__ src, unsigned short* __restrict__ dst,
                   int K, int N) {
    __shared__ float tile[32][33];
    int k0 = blockIdx.x * 32, n0 = blockIdx.y * 32;
    int tx = threadIdx.x & 31, ty = threadIdx.x >> 5;
    #pragma unroll
    for (int i = 0; i < 4; ++i) {
        int k = k0 + ty + i * 8;
        int n = n0 + tx;
        tile[ty + i * 8][tx] = (n < N) ? src[(size_t)k * N + n] : 0.f;
    }
    __syncthreads();
    #pragma unroll
    for (int i = 0; i < 4; ++i) {
        int n = n0 + ty + i * 8;
        dst[(size_t)n * K + k0 + tx] = f32_to_bf16(tile[tx][ty + i * 8]);
    }
}

// ---------------------------------------------------------------------------
// Plain fp32 -> bf16 convert
// ---------------------------------------------------------------------------
__global__ void cvt_kernel(const float* __restrict__ src,
                           unsigned short* __restrict__ dst, int n) {
    int i = blockIdx.x * blockDim.x + threadIdx.x;
    if (i < n) dst[i] = f32_to_bf16(src[i]);
}

// ---------------------------------------------------------------------------
// Embedding gather: h[t,:] = embed[ids[t],:]   (fp32 residual stream)
// ---------------------------------------------------------------------------
__global__ void gather_kernel(const int* __restrict__ ids,
                              const float* __restrict__ embed,
                              float* __restrict__ h) {
    int t = blockIdx.x;
    int id = ids[t];
    const float* src = embed + (size_t)id * D_MODEL;
    float* dst = h + (size_t)t * D_MODEL;
    for (int i = threadIdx.x; i < D_MODEL; i += blockDim.x) dst[i] = src[i];
}

// ---------------------------------------------------------------------------
// RMSNorm: fp32 in, bf16 out.  One block per row of 4096.
// ---------------------------------------------------------------------------
__global__ __launch_bounds__(256)
void rmsnorm_kernel(const float* __restrict__ x, const float* __restrict__ w,
                    unsigned short* __restrict__ y) {
    __shared__ float red[256];
    int row = blockIdx.x, tid = threadIdx.x;
    const float* xr = x + (size_t)row * D_MODEL;
    float ss = 0.f;
    for (int i = tid; i < D_MODEL; i += 256) { float t = xr[i]; ss += t * t; }
    red[tid] = ss;
    __syncthreads();
    for (int s = 128; s > 0; s >>= 1) {
        if (tid < s) red[tid] += red[tid + s];
        __syncthreads();
    }
    float inv = rsqrtf(red[0] / (float)D_MODEL + 1e-6f);
    unsigned short* yr = y + (size_t)row * D_MODEL;
    for (int i = tid; i < D_MODEL; i += 256) yr[i] = f32_to_bf16(xr[i] * inv * w[i]);
}

// ---------------------------------------------------------------------------
// RoPE in-place on bf16 q and k.  grid (S, H), 64 threads.
// ---------------------------------------------------------------------------
__global__ void rope_kernel(unsigned short* __restrict__ q, unsigned short* __restrict__ k) {
    int t = blockIdx.x, h = blockIdx.y, d = threadIdx.x;  // d in [0,64)
    float inv = __expf(-9.210340371976184f * (float)(2 * d) / (float)HEAD_DIM);
    float ang = (float)t * inv;
    float c = cosf(ang), s = sinf(ang);
    size_t base = (size_t)t * D_MODEL + (size_t)h * HEAD_DIM;
    float q0 = bf16_to_f32(q[base + d]), q1 = bf16_to_f32(q[base + d + 64]);
    q[base + d]      = f32_to_bf16(q0 * c - q1 * s);
    q[base + d + 64] = f32_to_bf16(q1 * c + q0 * s);
    float k0 = bf16_to_f32(k[base + d]), k1 = bf16_to_f32(k[base + d + 64]);
    k[base + d]      = f32_to_bf16(k0 * c - k1 * s);
    k[base + d + 64] = f32_to_bf16(k1 * c + k0 * s);
}

// ---------------------------------------------------------------------------
// Gate product on bf16: g = g * u
// ---------------------------------------------------------------------------
__global__ void mul_kernel(unsigned short* __restrict__ g,
                           const unsigned short* __restrict__ u, int n) {
    int i = blockIdx.x * blockDim.x + threadIdx.x;
    if (i < n) g[i] = f32_to_bf16(bf16_to_f32(g[i]) * bf16_to_f32(u[i]));
}

// ---------------------------------------------------------------------------
// Host orchestration
// ---------------------------------------------------------------------------
extern "C" void kernel_launch(void* const* d_in, const int* in_sizes, int n_in,
                              void* d_out, int out_size, void* d_ws, size_t ws_size,
                              hipStream_t stream) {
    const int*   ids    = (const int*)  d_in[0];
    const float* memory = (const float*)d_in[1];
    const float* embed  = (const float*)d_in[2];
    const float* lmhead = (const float*)d_in[3];
    const float* wq  = (const float*)d_in[4];
    const float* wk  = (const float*)d_in[5];
    const float* wv  = (const float*)d_in[6];
    const float* wo  = (const float*)d_in[7];
    const float* cwq = (const float*)d_in[8];
    const float* cwk = (const float*)d_in[9];
    const float* cwv = (const float*)d_in[10];
    const float* cwo = (const float*)d_in[11];
    const float* wg  = (const float*)d_in[12];
    const float* wu  = (const float*)d_in[13];
    const float* wd  = (const float*)d_in[14];
    const float* ln1 = (const float*)d_in[15];
    const float* lnc = (const float*)d_in[16];
    const float* ln2 = (const float*)d_in[17];
    const float* lnf = (const float*)d_in[18];
    const float* pw1 = (const float*)d_in[19];
    const float* pb1 = (const float*)d_in[20];
    const float* pw2 = (const float*)d_in[21];
    const float* pb2 = (const float*)d_in[22];

    const size_t SZ_D  = (size_t)S_LEN * D_MODEL;
    const size_t SZ_PH = (size_t)S_LEN * PH_DIM;
    const size_t SZ_FF = (size_t)S_LEN * FF_DIM;

    float* h = (float*)d_ws;                              // fp32 residual stream
    unsigned short* bb = (unsigned short*)(h + SZ_D);     // bf16 region
    unsigned short* memin  = bb;             bb += (size_t)S_LEN * DM_MEM;
    unsigned short* mem_h1 = bb;             bb += SZ_PH;
    unsigned short* mem    = bb;             bb += SZ_D;
    unsigned short* x      = bb;             bb += SZ_D;
    unsigned short* q      = bb;             bb += SZ_D;
    unsigned short* k      = bb;             bb += SZ_D;
    unsigned short* v      = bb;             bb += SZ_D;
    unsigned short* attn   = bb;             bb += SZ_D;
    unsigned short* g      = bb;             bb += SZ_FF;
    unsigned short* u      = bb;             bb += SZ_FF;
    unsigned short* wslot  = bb;             // rotating transposed-weight slot (45.1M el)

    auto cvtW = [&](const float* w, int K, int N, int Npad) {
        dim3 grid(K / 32, Npad / 32);
        cvt_wT_kernel<<<grid, 256, 0, stream>>>(w, wslot, K, N);
    };
    auto gemm = [&](const unsigned short* A, const float* bias, const float* resid,
                    void* C, int N, int Npad, int K, int act, bool obf16) {
        dim3 grid(Npad / 128, S_LEN / 128);
        if (obf16) {
            if (act == 1)
                gemm_bf16_wmma<1, true><<<grid, 256, 0, stream>>>(A, wslot, bias, resid, C, S_LEN, N, K);
            else if (act == 2)
                gemm_bf16_wmma<2, true><<<grid, 256, 0, stream>>>(A, wslot, bias, resid, C, S_LEN, N, K);
            else
                gemm_bf16_wmma<0, true><<<grid, 256, 0, stream>>>(A, wslot, bias, resid, C, S_LEN, N, K);
        } else {
            gemm_bf16_wmma<0, false><<<grid, 256, 0, stream>>>(A, wslot, bias, resid, C, S_LEN, N, K);
        }
    };

    // ---- projector: mem = gelu(memory @ p_w1 + b1) @ p_w2 + b2 ----
    cvt_kernel<<<(S_LEN * DM_MEM + 255) / 256, 256, 0, stream>>>(memory, memin, S_LEN * DM_MEM);
    cvtW(pw1, DM_MEM, PH_DIM, PH_DIM);
    gemm(memin, pb1, nullptr, mem_h1, PH_DIM, PH_DIM, DM_MEM, 1, true);
    cvtW(pw2, PH_DIM, D_MODEL, D_MODEL);
    gemm(mem_h1, pb2, nullptr, mem, D_MODEL, D_MODEL, PH_DIM, 0, true);

    // ---- embed ----
    gather_kernel<<<S_LEN, 256, 0, stream>>>(ids, embed, h);

    // ---- self-attention ----
    rmsnorm_kernel<<<S_LEN, 256, 0, stream>>>(h, ln1, x);
    cvtW(wq, D_MODEL, D_MODEL, D_MODEL); gemm(x, nullptr, nullptr, q, D_MODEL, D_MODEL, D_MODEL, 0, true);
    cvtW(wk, D_MODEL, D_MODEL, D_MODEL); gemm(x, nullptr, nullptr, k, D_MODEL, D_MODEL, D_MODEL, 0, true);
    cvtW(wv, D_MODEL, D_MODEL, D_MODEL); gemm(x, nullptr, nullptr, v, D_MODEL, D_MODEL, D_MODEL, 0, true);
    rope_kernel<<<dim3(S_LEN, N_HEADS), 64, 0, stream>>>(q, k);
    attn_wmma_kernel<<<dim3(S_LEN / 32, N_HEADS), 64, 0, stream>>>(q, k, v, attn);
    cvtW(wo, D_MODEL, D_MODEL, D_MODEL); gemm(attn, nullptr, h, h, D_MODEL, D_MODEL, D_MODEL, 0, false);

    // ---- cross-attention (causal) ----
    rmsnorm_kernel<<<S_LEN, 256, 0, stream>>>(h, lnc, x);
    cvtW(cwq, D_MODEL, D_MODEL, D_MODEL); gemm(x,   nullptr, nullptr, q, D_MODEL, D_MODEL, D_MODEL, 0, true);
    cvtW(cwk, D_MODEL, D_MODEL, D_MODEL); gemm(mem, nullptr, nullptr, k, D_MODEL, D_MODEL, D_MODEL, 0, true);
    cvtW(cwv, D_MODEL, D_MODEL, D_MODEL); gemm(mem, nullptr, nullptr, v, D_MODEL, D_MODEL, D_MODEL, 0, true);
    attn_wmma_kernel<<<dim3(S_LEN / 32, N_HEADS), 64, 0, stream>>>(q, k, v, attn);
    cvtW(cwo, D_MODEL, D_MODEL, D_MODEL); gemm(attn, nullptr, h, h, D_MODEL, D_MODEL, D_MODEL, 0, false);

    // ---- SwiGLU MLP ----
    rmsnorm_kernel<<<S_LEN, 256, 0, stream>>>(h, ln2, x);
    cvtW(wg, D_MODEL, FF_DIM, FF_DIM); gemm(x, nullptr, nullptr, g, FF_DIM, FF_DIM, D_MODEL, 2, true);
    cvtW(wu, D_MODEL, FF_DIM, FF_DIM); gemm(x, nullptr, nullptr, u, FF_DIM, FF_DIM, D_MODEL, 0, true);
    mul_kernel<<<((int)SZ_FF + 255) / 256, 256, 0, stream>>>(g, u, (int)SZ_FF);
    cvtW(wd, FF_DIM, D_MODEL, D_MODEL); gemm(g, nullptr, h, h, D_MODEL, D_MODEL, FF_DIM, 0, false);

    // ---- final norm + LM head ----
    rmsnorm_kernel<<<S_LEN, 256, 0, stream>>>(h, lnf, x);
    cvtW(lmhead, D_MODEL, VOCAB, VOCAB_PAD);
    gemm(x, nullptr, nullptr, (float*)d_out, VOCAB, VOCAB_PAD, D_MODEL, 0, false);
}
